// GIN_33492154974257
// MI455X (gfx1250) — matrix operand order repaired and checked by reading, not compiled, not run on hardware
//
#include <hip/hip_runtime.h>
#include <hip/hip_bf16.h>

typedef __attribute__((ext_vector_type(16))) _Float16 v16h;
typedef __attribute__((ext_vector_type(8)))  _Float16 v8h;
typedef __attribute__((ext_vector_type(8)))  float    v8f;

#define N_NODES    100000
#define N_EDGES    1600000
#define NUM_GRAPHS 1024
#define POOL_W     448
#define BN_EPS     1e-5f

// ---------------- utility kernels ----------------

__global__ void gin_zero_f(float* p, int n) {
    int i = blockIdx.x * blockDim.x + threadIdx.x;
    if (i < n) p[i] = 0.0f;
}

__global__ void gin_copy_f4(const float4* __restrict__ src, float4* __restrict__ dst, int n4) {
    int i = blockIdx.x * blockDim.x + threadIdx.x;
    if (i < n4) dst[i] = src[i];
}

// scatter-add aggregation: agg[dst] += h[src], vector-of-4 per thread
__global__ void gin_edge_agg(const float* __restrict__ h,
                             const int* __restrict__ src,
                             const int* __restrict__ dst,
                             float* __restrict__ agg,
                             int C, int total) {
    int i = blockIdx.x * blockDim.x + threadIdx.x;
    if (i >= total) return;
    int chunks = C >> 2;
    int e  = i / chunks;
    int c4 = (i - e * chunks) << 2;
    int s = src[e];
    int d = dst[e];
    const float4 v = *(const float4*)(h + (size_t)s * C + c4);
    float* out = agg + (size_t)d * C + c4;
    atomicAdd(out + 0, v.x);
    atomicAdd(out + 1, v.y);
    atomicAdd(out + 2, v.z);
    atomicAdd(out + 3, v.w);
}

// one-time split of fp32 -> (hi, lo) f16 pair, elementwise
__global__ void gin_split(const float* __restrict__ src, _Float16* __restrict__ hi,
                          _Float16* __restrict__ lo, int n) {
    int i = blockIdx.x * blockDim.x + threadIdx.x;
    if (i >= n) return;
    float f = src[i];
    _Float16 h = (_Float16)f;
    hi[i] = h;
    lo[i] = (_Float16)(f - (float)h);
}

// split fp32 W[K,N] -> transposed (hi, lo) f16 WT[N,K]
__global__ void gin_split_tr(const float* __restrict__ W, _Float16* __restrict__ WTh,
                             _Float16* __restrict__ WTl, int K, int N) {
    int i = blockIdx.x * blockDim.x + threadIdx.x;
    if (i >= K * N) return;
    int k = i / N;
    int n = i - k * N;
    float f = W[i];
    _Float16 h = (_Float16)f;
    WTh[(size_t)n * K + k] = h;
    WTl[(size_t)n * K + k] = (_Float16)(f - (float)h);
}

// ---------------- WMMA fragment loaders (pre-split f16 data) ----------------

// A 16x32 f16 fragment: lane L holds row (L&15),
// halves j = K[kb+j], kb = k0 + 16*(L>=16)  -> 16 contiguous halves
__device__ __forceinline__ v16h gin_load_a(const _Float16* __restrict__ A, int K,
                                           int m, int kb) {
    const _Float16* p = A + (size_t)m * K + kb;
    v8h a0 = *(const v8h*)(p);
    v8h a1 = *(const v8h*)(p + 8);
    return __builtin_shufflevector(a0, a1, 0, 1, 2, 3, 4, 5, 6, 7,
                                           8, 9, 10, 11, 12, 13, 14, 15);
}

// B 32x16 f16 fragment from transposed weights WT[N,K]: lane L holds col (L&15),
// halves j = K[kb2+j], kb2 = k0 + 16*(L>=16)  -> 16 contiguous halves
__device__ __forceinline__ v16h gin_load_b(const _Float16* __restrict__ WT, int K,
                                           int n, int kb2) {
    return *(const v16h*)(WT + (size_t)n * K + kb2);
}

__device__ __forceinline__ v8f gin_mm_step(v8f acc, v16h ahi, v16h alo,
                                           v16h bhi, v16h blo) {
    acc = __builtin_amdgcn_wmma_f32_16x16x32_f16(false, ahi, false, bhi, (short)0, acc, false, false);
    acc = __builtin_amdgcn_wmma_f32_16x16x32_f16(false, ahi, false, blo, (short)0, acc, false, false);
    acc = __builtin_amdgcn_wmma_f32_16x16x32_f16(false, alo, false, bhi, (short)0, acc, false, false);
    return acc;
}

// ---------------- GEMM + bias + BN + ReLU; writes (hi, lo) f16 output ----------------
// One wave per 16x16 tile; 8 waves per block. M%16==0, N%16==0, K%32==0.
__global__ __launch_bounds__(256) void gin_gemm_bn_relu(
    const _Float16* __restrict__ Ah, const _Float16* __restrict__ Al,
    const _Float16* __restrict__ WTh, const _Float16* __restrict__ WTl,
    const float* __restrict__ bias, const float* __restrict__ g,
    const float* __restrict__ bta, const float* __restrict__ rm,
    const float* __restrict__ rv,
    _Float16* __restrict__ outh, _Float16* __restrict__ outl,
    int M, int K, int N) {
    int lane = threadIdx.x & 31;
    int wave = threadIdx.x >> 5;
    int mtiles = M >> 4;
    int ntiles = N >> 4;
    int tile = blockIdx.x * 8 + wave;
    if (tile >= mtiles * ntiles) return;
    int tm = tile % mtiles;
    int tn = tile / mtiles;
    int m = (tm << 4) + (lane & 15);
    int n = (tn << 4) + (lane & 15);
    int ka = (lane >> 4) << 4;   // +16 for upper half-wave (A)
    int kb = (lane >> 4) << 4;   // +16 for upper half-wave (B)

    v8f acc = {};
    for (int k0 = 0; k0 < K; k0 += 32) {
        v16h ahi = gin_load_a(Ah, K, m, k0 + ka);
        v16h alo = gin_load_a(Al, K, m, k0 + ka);
        v16h bhi = gin_load_b(WTh, K, n, k0 + kb);
        v16h blo = gin_load_b(WTl, K, n, k0 + kb);
        acc = gin_mm_step(acc, ahi, alo, bhi, blo);
    }

    float bv  = bias[n];
    float sc  = g[n] / sqrtf(rv[n] + BN_EPS);
    float rmv = rm[n];
    float btv = bta[n];
    int row0 = (tm << 4) + ((lane >> 4) << 3);
#pragma unroll
    for (int r = 0; r < 8; ++r) {
        float v = acc[r] + bv;
        v = (v - rmv) * sc + btv;
        v = v > 0.0f ? v : 0.0f;
        size_t idx = (size_t)(row0 + r) * N + n;
        _Float16 h = (_Float16)v;
        outh[idx] = h;
        outl[idx] = (_Float16)(v - (float)h);
    }
}

// ---------------- GEMM + bias + ReLU + fused graph pooling (f32 h out) ----------------
__global__ __launch_bounds__(256) void gin_gemm_relu_pool(
    const _Float16* __restrict__ Ah, const _Float16* __restrict__ Al,
    const _Float16* __restrict__ WTh, const _Float16* __restrict__ WTl,
    const float* __restrict__ bias, float* __restrict__ h_out,
    const int* __restrict__ batch, float* __restrict__ pooled,
    int M, int K, int N, int pool_off, int store_h) {
    int lane = threadIdx.x & 31;
    int wave = threadIdx.x >> 5;
    int mtiles = M >> 4;
    int ntiles = N >> 4;
    int tile = blockIdx.x * 8 + wave;
    if (tile >= mtiles * ntiles) return;
    int tm = tile % mtiles;
    int tn = tile / mtiles;
    int m = (tm << 4) + (lane & 15);
    int n = (tn << 4) + (lane & 15);
    int ka = (lane >> 4) << 4;
    int kb = (lane >> 4) << 4;

    v8f acc = {};
    for (int k0 = 0; k0 < K; k0 += 32) {
        v16h ahi = gin_load_a(Ah, K, m, k0 + ka);
        v16h alo = gin_load_a(Al, K, m, k0 + ka);
        v16h bhi = gin_load_b(WTh, K, n, k0 + kb);
        v16h blo = gin_load_b(WTl, K, n, k0 + kb);
        acc = gin_mm_step(acc, ahi, alo, bhi, blo);
    }

    float bv = bias[n];
    int row0 = (tm << 4) + ((lane >> 4) << 3);
#pragma unroll
    for (int r = 0; r < 8; ++r) {
        int row = row0 + r;
        float v = acc[r] + bv;
        v = v > 0.0f ? v : 0.0f;
        if (store_h) h_out[(size_t)row * N + n] = v;
        int gidx = batch[row];
        atomicAdd(pooled + (size_t)gidx * POOL_W + pool_off + n, v);
    }
}

// ---------------- final tiny layer: out = (m1h+m1l) @ W + b ----------------
__global__ void gin_mlp_last(const _Float16* __restrict__ inh, const _Float16* __restrict__ inl,
                             const float* __restrict__ W, const float* __restrict__ b,
                             float* __restrict__ out, int K, int N) {
    int i = blockIdx.x * blockDim.x + threadIdx.x;
    if (i >= NUM_GRAPHS * N) return;
    int row = i / N;
    int n = i - row * N;
    float s = b[n];
    for (int k = 0; k < K; ++k) {
        float a = (float)inh[(size_t)row * K + k] + (float)inl[(size_t)row * K + k];
        s += a * W[(size_t)k * N + n];
    }
    out[i] = s;
}

// ---------------- host orchestration ----------------

extern "C" void kernel_launch(void* const* d_in, const int* in_sizes, int n_in,
                              void* d_out, int out_size, void* d_ws, size_t ws_size,
                              hipStream_t stream) {
    const float* x     = (const float*)d_in[0];
    const int*   edges = (const int*)d_in[1];
    const int*   batch = (const int*)d_in[2];
    const int* e_src = edges;
    const int* e_dst = edges + N_EDGES;

    // conv params: 3 layers x {W1,b1,g,bta,rm,rv,W2,b2} starting at index 3
    const float* cp[3][8];
    for (int l = 0; l < 3; ++l)
        for (int j = 0; j < 8; ++j)
            cp[l][j] = (const float*)d_in[3 + l * 8 + j];
    // mlp params: layers 0..3 {W,b,g,bta,rm,rv} at 27..50, layer 4 {W,b} at 51..52
    const float* mp[5][6];
    for (int l = 0; l < 4; ++l)
        for (int j = 0; j < 6; ++j)
            mp[l][j] = (const float*)d_in[27 + l * 6 + j];
    mp[4][0] = (const float*)d_in[51];
    mp[4][1] = (const float*)d_in[52];

    // ---------------- workspace layout (byte offsets, 256B aligned) ----------------
    char* base = (char*)d_ws;
    size_t off = 0;
    auto alloc = [&](size_t bytes) -> char* {
        char* p = base + off;
        off += (bytes + 255) & ~(size_t)255;
        return p;
    };
    float*    agg     = (float*)   alloc((size_t)N_NODES * 128 * 4);
    _Float16* aggh    = (_Float16*)alloc((size_t)N_NODES * 128 * 2);
    _Float16* aggl    = (_Float16*)alloc((size_t)N_NODES * 128 * 2);
    _Float16* th      = (_Float16*)alloc((size_t)N_NODES * 256 * 2);
    _Float16* tl      = (_Float16*)alloc((size_t)N_NODES * 256 * 2);
    float*    hA      = (float*)   alloc((size_t)N_NODES * 64 * 4);
    float*    hB      = (float*)   alloc((size_t)N_NODES * 128 * 4);
    float*    pooled  = (float*)   alloc((size_t)NUM_GRAPHS * POOL_W * 4);
    _Float16* pooledh = (_Float16*)alloc((size_t)NUM_GRAPHS * POOL_W * 2);
    _Float16* pooledl = (_Float16*)alloc((size_t)NUM_GRAPHS * POOL_W * 2);
    _Float16* m0h     = (_Float16*)alloc((size_t)NUM_GRAPHS * 64 * 2);
    _Float16* m0l     = (_Float16*)alloc((size_t)NUM_GRAPHS * 64 * 2);
    _Float16* m1h     = (_Float16*)alloc((size_t)NUM_GRAPHS * 64 * 2);
    _Float16* m1l     = (_Float16*)alloc((size_t)NUM_GRAPHS * 64 * 2);

    const int ci[3] = {128, 64, 128};   // conv input widths
    const int hi[3] = {64, 128, 256};   // conv output widths
    // transposed/split weights
    _Float16 *w1h[3], *w1l[3], *w2h[3], *w2l[3], *mwh[4], *mwl[4];
    for (int l = 0; l < 3; ++l) {
        w1h[l] = (_Float16*)alloc((size_t)ci[l] * hi[l] * 2);
        w1l[l] = (_Float16*)alloc((size_t)ci[l] * hi[l] * 2);
        w2h[l] = (_Float16*)alloc((size_t)hi[l] * hi[l] * 2);
        w2l[l] = (_Float16*)alloc((size_t)hi[l] * hi[l] * 2);
    }
    const int mk[4] = {POOL_W, 64, 64, 64};
    for (int l = 0; l < 4; ++l) {
        mwh[l] = (_Float16*)alloc((size_t)mk[l] * 64 * 2);
        mwl[l] = (_Float16*)alloc((size_t)mk[l] * 64 * 2);
    }

    // ---------------- one-time weight split+transpose ----------------
    for (int l = 0; l < 3; ++l) {
        int n1 = ci[l] * hi[l], n2 = hi[l] * hi[l];
        gin_split_tr<<<(n1 + 255) / 256, 256, 0, stream>>>(cp[l][0], w1h[l], w1l[l], ci[l], hi[l]);
        gin_split_tr<<<(n2 + 255) / 256, 256, 0, stream>>>(cp[l][6], w2h[l], w2l[l], hi[l], hi[l]);
    }
    for (int l = 0; l < 4; ++l) {
        int n1 = mk[l] * 64;
        gin_split_tr<<<(n1 + 255) / 256, 256, 0, stream>>>(mp[l][0], mwh[l], mwl[l], mk[l], 64);
    }

    // zero pooled accumulator
    {
        int n = NUM_GRAPHS * POOL_W;
        gin_zero_f<<<(n + 255) / 256, 256, 0, stream>>>(pooled, n);
    }

    const int poff[3] = {0, 64, 192};
    const float* h_in[3] = {x, hA, hB};
    float* h_next[3] = {hA, hB, nullptr};

    for (int l = 0; l < 3; ++l) {
        int C = ci[l], H = hi[l];
        // agg = h_in (self term, eps = 0)
        {
            int n4 = N_NODES * C / 4;
            gin_copy_f4<<<(n4 + 255) / 256, 256, 0, stream>>>((const float4*)h_in[l], (float4*)agg, n4);
        }
        // agg[dst] += h[src]
        {
            int total = N_EDGES * (C / 4);
            gin_edge_agg<<<(total + 255) / 256, 256, 0, stream>>>(h_in[l], e_src, e_dst, agg, C, total);
        }
        // split agg -> f16 hi/lo
        {
            int n = N_NODES * C;
            gin_split<<<(n + 255) / 256, 256, 0, stream>>>(agg, aggh, aggl, n);
        }
        // t = relu(bn(agg @ W1 + b1))  (written as hi/lo)
        {
            int tiles = (N_NODES >> 4) * (H >> 4);
            gin_gemm_bn_relu<<<(tiles + 7) / 8, 256, 0, stream>>>(
                aggh, aggl, w1h[l], w1l[l], cp[l][1], cp[l][2], cp[l][3], cp[l][4], cp[l][5],
                th, tl, N_NODES, C, H);
        }
        // h_next = relu(t @ W2 + b2); pooled[batch] += h_next (fused)
        {
            int tiles = (N_NODES >> 4) * (H >> 4);
            gin_gemm_relu_pool<<<(tiles + 7) / 8, 256, 0, stream>>>(
                th, tl, w2h[l], w2l[l], cp[l][7], h_next[l], batch, pooled,
                N_NODES, H, H, poff[l], h_next[l] != nullptr ? 1 : 0);
        }
    }

    // split pooled -> hi/lo
    {
        int n = NUM_GRAPHS * POOL_W;
        gin_split<<<(n + 255) / 256, 256, 0, stream>>>(pooled, pooledh, pooledl, n);
    }

    // head MLP: 448 -> 64 -> 64 -> 64 -> 64 -> 16
    _Float16* inh = pooledh; _Float16* inl = pooledl;
    _Float16* oh[4] = {m0h, m1h, m0h, m1h};
    _Float16* ol[4] = {m0l, m1l, m0l, m1l};
    int kdim = POOL_W;
    for (int l = 0; l < 4; ++l) {
        int tiles = (NUM_GRAPHS >> 4) * (64 >> 4);
        gin_gemm_bn_relu<<<(tiles + 7) / 8, 256, 0, stream>>>(
            inh, inl, mwh[l], mwl[l], mp[l][1], mp[l][2], mp[l][3], mp[l][4], mp[l][5],
            oh[l], ol[l], NUM_GRAPHS, kdim, 64);
        inh = oh[l]; inl = ol[l];
        kdim = 64;
    }
    {
        int n = NUM_GRAPHS * 16;
        gin_mlp_last<<<(n + 255) / 256, 256, 0, stream>>>(m1h, m1l, mp[4][0], mp[4][1],
                                                          (float*)d_out, 64, 16);
    }
}
